// GridLSTM_50345606643940
// MI455X (gfx1250) — matrix operand order repaired and checked by reading, not compiled, not run on hardware
//
#include <hip/hip_runtime.h>
#include <hip/hip_bf16.h>

// ---------------- problem constants ----------------
#define Bn   16
#define Cn   64
#define Hn   128
#define G1   64
#define G2   64
#define GATE 512            // 4*H per dim
#define HP   264            // LDS pitch for Hcat (256 + 8 halfs pad)
#define XP   72             // LDS pitch for x tile (64 + 8 halfs pad)

typedef __attribute__((ext_vector_type(16))) __bf16 v16bf;
typedef __attribute__((ext_vector_type(8)))  float  v8f;

union Frag {                 // 32-byte WMMA operand (16 bf16)
    uint4 q[2];
    v16bf v;
};

__device__ __forceinline__ unsigned short f32_to_bf16(float f) {
    union { float f; unsigned u; } v; v.f = f;
    unsigned u = v.u;
    u += 0x7FFFu + ((u >> 16) & 1u);   // round-to-nearest-even
    return (unsigned short)(u >> 16);
}

__device__ __forceinline__ float sigmoidf_fast(float x) {
    return 1.0f / (1.0f + __expf(-x));
}

// ---------------- preprocessing kernels ----------------

// x[b][c][i][j] (f32) -> xr[i][j][b][c] (bf16)
__global__ void reorder_x_kernel(const float* __restrict__ x,
                                 unsigned short* __restrict__ xr) {
    int o = blockIdx.x * blockDim.x + threadIdx.x;
    if (o >= Bn * Cn * G1 * G2) return;
    int c = o & 63;
    int b = (o >> 6) & 15;
    int j = (o >> 10) & 63;
    int i = (o >> 16) & 63;
    float v = x[(((b * Cn + c) * G1 + i) * G2) + j];
    xr[o] = f32_to_bf16(v);
}

// W_ih[d][n][k] (f32, n<512,k<64) -> fragment-major bf16:
// idx = ((((d*32 + t)*2 + s)*32 + lane)*16 + e), n=t*16+(lane&15), k=s*32+(lane>>4)*16+e
__global__ void pack_wih_kernel(const float* __restrict__ W,
                                unsigned short* __restrict__ out) {
    int u = blockIdx.x * blockDim.x + threadIdx.x;
    if (u >= 2 * 32 * 2 * 32 * 16) return;
    int e    = u & 15;
    int lane = (u >> 4) & 31;
    int s    = (u >> 9) & 1;
    int t    = (u >> 10) & 31;
    int d    = (u >> 15) & 1;
    int n = t * 16 + (lane & 15);
    int k = s * 32 + (lane >> 4) * 16 + e;
    out[u] = f32_to_bf16(W[(d * GATE + n) * Cn + k]);
}

// W_hh[d][n][k] (f32, n<512,k<256) -> fragment-major bf16:
// idx = ((((d*32 + t)*8 + s)*32 + lane)*16 + e)
__global__ void pack_whh_kernel(const float* __restrict__ W,
                                unsigned short* __restrict__ out) {
    int u = blockIdx.x * blockDim.x + threadIdx.x;
    if (u >= 2 * 32 * 8 * 32 * 16) return;
    int e    = u & 15;
    int lane = (u >> 4) & 31;
    int s    = (u >> 9) & 7;
    int t    = (u >> 12) & 31;
    int d    = (u >> 17) & 1;
    int n = t * 16 + (lane & 15);
    int k = s * 32 + (lane >> 4) * 16 + e;
    out[u] = f32_to_bf16(W[(d * GATE + n) * (2 * Hn) + k]);
}

__global__ void make_bias_kernel(const float* __restrict__ b_ih,
                                 const float* __restrict__ b_hh,
                                 float* __restrict__ bias) {
    int u = blockIdx.x * blockDim.x + threadIdx.x;
    if (u < 2 * GATE) bias[u] = b_ih[u] + b_hh[u];
}

// ---------------- wavefront cell kernel ----------------
// One workgroup (8 waves) per grid cell on the current anti-diagonal.
// Each wave owns 2 tasks (dim d, hidden-tile ht); per task it keeps 4 f32
// accumulators (gates i,f,g,o) live and finishes the LSTM update in registers.
__global__ __launch_bounds__(256) void grid_lstm_cell_kernel(
    const unsigned short* __restrict__ xr,     // [i][j][b][c] bf16
    const unsigned short* __restrict__ wih_f,  // fragment-major bf16
    const unsigned short* __restrict__ whh_f,  // fragment-major bf16
    const float* __restrict__ bias,            // [2][512] f32 (b_ih+b_hh)
    float* __restrict__ h1s, float* __restrict__ c1s,   // [G2][B][H] dim-0 state
    float* __restrict__ h2s, float* __restrict__ c2s,   // [G1][B][H] dim-1 state
    float* __restrict__ out,                   // [B][2H][G1][G2]
    int diag, int ilo)
{
    __shared__ unsigned short lds_h[16 * HP];  // Hcat [b][2H] bf16 (padded)
    __shared__ unsigned short lds_x[16 * XP];  // x_ij [b][C]  bf16 (padded)

    const int i = ilo + blockIdx.x;
    const int j = diag - i;
    const int tid = threadIdx.x;

    // Stage Hcat = [h_up | h_left] into LDS (f32 -> bf16); zeros at boundary.
    for (int e = tid; e < 16 * 256; e += 256) {
        int b = e >> 8, k = e & 255;
        float v;
        if (k < Hn) v = (i > 0) ? h1s[(j * Bn + b) * Hn + k]        : 0.0f;
        else        v = (j > 0) ? h2s[(i * Bn + b) * Hn + (k - Hn)] : 0.0f;
        lds_h[b * HP + k] = f32_to_bf16(v);
    }
    // Stage x_ij (already bf16)
    for (int e = tid; e < 16 * 64; e += 256) {
        int b = e >> 6, c = e & 63;
        lds_x[b * XP + c] = xr[((i * G2 + j) * Bn + b) * Cn + c];
    }
    __syncthreads();

    const int wave = tid >> 5;
    const int lane = tid & 31;
    const int m  = lane & 15;     // A-matrix row (batch)
    const int g  = lane >> 4;     // lane-group for K split
    const int nl = lane & 15;     // B/C/D column within tile

    for (int task = wave; task < 16; task += 8) {
        const int d  = task >> 3;      // grid dimension 0/1
        const int ht = task & 7;       // hidden tile (16 units)

        v8f acc[4];
        for (int gt = 0; gt < 4; ++gt) {
            float bv = bias[d * GATE + gt * Hn + ht * 16 + nl];
            v8f a;
            for (int r = 0; r < 8; ++r) a[r] = bv;
            acc[gt] = a;
        }

        // input contribution: K = 64 -> 2 WMMA k-steps
        for (int s = 0; s < 2; ++s) {
            Frag a;
            a.q[0] = *(const uint4*)&lds_x[m * XP + s * 32 + g * 16];
            a.q[1] = *(const uint4*)&lds_x[m * XP + s * 32 + g * 16 + 8];
            for (int gt = 0; gt < 4; ++gt) {
                const int t = gt * 8 + ht;
                Frag bf;
                const uint4* bp = (const uint4*)&wih_f[(((d * 32 + t) * 2 + s) * 32 + lane) * 16];
                bf.q[0] = bp[0]; bf.q[1] = bp[1];
                acc[gt] = __builtin_amdgcn_wmma_f32_16x16x32_bf16(
                    false, a.v, false, bf.v, (short)0, acc[gt], false, false);
            }
        }
        // recurrent contribution: K = 256 -> 8 WMMA k-steps
        for (int s = 0; s < 8; ++s) {
            Frag a;
            a.q[0] = *(const uint4*)&lds_h[m * HP + s * 32 + g * 16];
            a.q[1] = *(const uint4*)&lds_h[m * HP + s * 32 + g * 16 + 8];
            for (int gt = 0; gt < 4; ++gt) {
                const int t = gt * 8 + ht;
                Frag bf;
                const uint4* bp = (const uint4*)&whh_f[(((d * 32 + t) * 8 + s) * 32 + lane) * 16];
                bf.q[0] = bp[0]; bf.q[1] = bp[1];
                acc[gt] = __builtin_amdgcn_wmma_f32_16x16x32_bf16(
                    false, a.v, false, bf.v, (short)0, acc[gt], false, false);
            }
        }

        // LSTM pointwise update fully in registers (C/D layout:
        // vgpr r, lanes 0-15 -> batch r; lanes 16-31 -> batch r+8; col = lane&15)
        const int h_idx = ht * 16 + nl;
        float* cs = d ? c2s : c1s;
        float* hs = d ? h2s : h1s;
        const int pos = d ? i : j;
        const bool has_c = d ? (j > 0) : (i > 0);
        for (int r = 0; r < 8; ++r) {
            const int b = r + g * 8;
            const int sidx = (pos * Bn + b) * Hn + h_idx;
            float cprev = has_c ? cs[sidx] : 0.0f;
            float gi = acc[0][r], gf = acc[1][r], gg = acc[2][r], go = acc[3][r];
            float c = sigmoidf_fast(gf) * cprev + sigmoidf_fast(gi) * tanhf(gg);
            float h = sigmoidf_fast(go) * tanhf(c);
            cs[sidx] = c;
            hs[sidx] = h;
            out[(((size_t)b * (2 * Hn) + d * Hn + h_idx) * G1 + i) * G2 + j] = h;
        }
    }
}

// ---------------- launch ----------------
extern "C" void kernel_launch(void* const* d_in, const int* in_sizes, int n_in,
                              void* d_out, int out_size, void* d_ws, size_t ws_size,
                              hipStream_t stream) {
    (void)in_sizes; (void)n_in; (void)out_size; (void)ws_size;
    const float* x    = (const float*)d_in[0];
    const float* W_ih = (const float*)d_in[1];
    const float* W_hh = (const float*)d_in[2];
    const float* b_ih = (const float*)d_in[3];
    const float* b_hh = (const float*)d_in[4];
    float* out = (float*)d_out;

    char* ws = (char*)d_ws;
    size_t off = 0;
    unsigned short* xr   = (unsigned short*)(ws + off); off += (size_t)G1 * G2 * Bn * Cn * 2;
    unsigned short* wihf = (unsigned short*)(ws + off); off += (size_t)2 * 32 * 2 * 32 * 16 * 2;
    unsigned short* whhf = (unsigned short*)(ws + off); off += (size_t)2 * 32 * 8 * 32 * 16 * 2;
    float* bias = (float*)(ws + off); off += (size_t)2 * GATE * 4;
    float* h1s  = (float*)(ws + off); off += (size_t)G2 * Bn * Hn * 4;
    float* c1s  = (float*)(ws + off); off += (size_t)G2 * Bn * Hn * 4;
    float* h2s  = (float*)(ws + off); off += (size_t)G1 * Bn * Hn * 4;
    float* c2s  = (float*)(ws + off); off += (size_t)G1 * Bn * Hn * 4;

    // preprocessing (bf16 conversion + fragment-major weight packing)
    {
        int n = Bn * Cn * G1 * G2;
        reorder_x_kernel<<<(n + 255) / 256, 256, 0, stream>>>(x, xr);
    }
    pack_wih_kernel<<<(2 * 32 * 2 * 32 * 16) / 256, 256, 0, stream>>>(W_ih, wihf);
    pack_whh_kernel<<<(2 * 32 * 8 * 32 * 16) / 256, 256, 0, stream>>>(W_hh, whhf);
    make_bias_kernel<<<4, 256, 0, stream>>>(b_ih, b_hh, bias);

    // 2-D wavefront: one launch per anti-diagonal, one workgroup per cell.
    // Boundary cells branch to zero state, so no state init is required.
    for (int dgl = 0; dgl < G1 + G2 - 1; ++dgl) {
        int ilo = dgl > (G2 - 1) ? dgl - (G2 - 1) : 0;
        int ihi = dgl < (G1 - 1) ? dgl : (G1 - 1);
        int ncells = ihi - ilo + 1;
        grid_lstm_cell_kernel<<<ncells, 256, 0, stream>>>(
            xr, wihf, whhf, bias, h1s, c1s, h2s, c2s, out, dgl, ilo);
    }
}
